// SwinTransformerBlock3D_3178275799716
// MI455X (gfx1250) — compile-verified, matrix operand order checked by reading, not run on hardware
//
#include <hip/hip_runtime.h>
#include <hip/hip_bf16.h>
#include <math.h>

// ---------------------------------------------------------------------------
// Swin-3D block on gfx1250 (MI455X): wave32 + v_wmma_f32_16x16x32_f16 path.
// ---------------------------------------------------------------------------

typedef _Float16 half_t;
typedef __attribute__((ext_vector_type(16))) _Float16 v16h;
typedef __attribute__((ext_vector_type(8)))  float    v8f;
typedef __attribute__((ext_vector_type(4)))  _Float16 h4;

#define T_TOKENS 200704          // 8*8*56*56
#define NWIN     2048            // windows total (8 * 256)
#define NTOK     98              // tokens per window
#define HEADS    4
#define HD       32
#define SCALE_Q  0.1767766952966369f  // 32^-0.5

// ---- WMMA wrapper ---------------------------------------------------------
__device__ __forceinline__ v8f wmma16(v16h a, v16h b, v8f c) {
  // D = A(16x32) x B(32x16) + C, fp32 accumulate
  return __builtin_amdgcn_wmma_f32_16x16x32_f16(false, a, false, b,
                                                (short)0, c, false, false);
}

// ---- fragment loads (ISA 7.12.2 16-bit A/B layout) ------------------------
// lane&15 selects row (A) / column (B); halves cover K = {0..7,16..23} for
// lanes 0-15 and {8..15,24..31} for lanes 16-31 -> two 16-byte chunks.
__device__ __forceinline__ v16h frag_load(const half_t* p, int ld, int lane) {
  const half_t* q = p + (size_t)(lane & 15) * ld + ((lane & 16) ? 8 : 0);
  v16h out;
  float4* o = reinterpret_cast<float4*>(&out);
  o[0] = *reinterpret_cast<const float4*>(q);
  o[1] = *reinterpret_cast<const float4*>(q + 16);
  return out;
}

__device__ __forceinline__ v16h frag_load_guard(const half_t* p, int ld,
                                                int valid, int lane) {
  int r = lane & 15;
  v16h out = {};
  if (r < valid) {
    const half_t* q = p + (size_t)r * ld + ((lane & 16) ? 8 : 0);
    float4* o = reinterpret_cast<float4*>(&out);
    o[0] = *reinterpret_cast<const float4*>(q);
    o[1] = *reinterpret_cast<const float4*>(q + 16);
  }
  return out;
}

// ---- window row (wi*98+t) -> flat spatial token index ---------------------
// Forward roll (-1,-3,-3) gathers xn[q+s]; reverse roll (+1,+3,+3) scatters
// back to q+s, so gather source and scatter destination coincide.
__device__ __forceinline__ int win_row_to_spatial(int r) {
  int wi = r / NTOK, t = r - wi * NTOK;
  int b = wi >> 8, rem = wi & 255;
  int dz = rem >> 6, hy = (rem >> 3) & 7, wx = rem & 7;
  int td = t / 49, tr = t - td * 49;
  int th = tr / 7,  tw = tr - th * 7;
  int zd = dz * 2 + td, yh = hy * 7 + th, xw = wx * 7 + tw;
  int z = (zd + 1) & 7;
  int y = yh + 3; if (y >= 56) y -= 56;
  int x = xw + 3; if (x >= 56) x -= 56;
  return ((b * 8 + z) * 56 + y) * 56 + x;
}

// ---------------------------------------------------------------------------
// Kernel 0: weight fp32 [K][N] -> fp16 transposed [N][K]
// ---------------------------------------------------------------------------
__global__ void wprep_kernel(const float* __restrict__ in, half_t* __restrict__ out,
                             int K, int N) {
  int idx = blockIdx.x * 256 + threadIdx.x;
  if (idx < K * N) {
    int k = idx / N, n = idx - k * N;
    out[(size_t)n * K + k] = (half_t)in[idx];
  }
}

// ---------------------------------------------------------------------------
// Kernel 1: LN1 + cyclic shift + 3D window partition -> xw fp16 [T,128]
// one wave (32 lanes x 4 ch) per token row
// ---------------------------------------------------------------------------
__global__ __launch_bounds__(256) void ln1_kernel(const float* __restrict__ x,
                                                  const float* __restrict__ g,
                                                  const float* __restrict__ b,
                                                  half_t* __restrict__ xw) {
  int lane = threadIdx.x & 31;
  int r = blockIdx.x * 8 + (threadIdx.x >> 5);
  int p = win_row_to_spatial(r);
  float4 xv = reinterpret_cast<const float4*>(x + (size_t)p * 128)[lane];
  float s = xv.x + xv.y + xv.z + xv.w;
  #pragma unroll
  for (int off = 16; off >= 1; off >>= 1) s += __shfl_xor(s, off, 32);
  float mu = s * (1.0f / 128.0f);
  float dx = xv.x - mu, dy = xv.y - mu, dz = xv.z - mu, dw = xv.w - mu;
  float v = dx * dx + dy * dy + dz * dz + dw * dw;
  #pragma unroll
  for (int off = 16; off >= 1; off >>= 1) v += __shfl_xor(v, off, 32);
  float rs = rsqrtf(v * (1.0f / 128.0f) + 1e-5f);
  float4 gv = reinterpret_cast<const float4*>(g)[lane];
  float4 bv = reinterpret_cast<const float4*>(b)[lane];
  h4 o;
  o[0] = (half_t)(dx * rs * gv.x + bv.x);
  o[1] = (half_t)(dy * rs * gv.y + bv.y);
  o[2] = (half_t)(dz * rs * gv.z + bv.z);
  o[3] = (half_t)(dw * rs * gv.w + bv.w);
  *reinterpret_cast<h4*>(xw + (size_t)r * 128 + lane * 4) = o;
}

// ---------------------------------------------------------------------------
// Kernel 2: QKV GEMM [T,128] x [128,384]^T(fp16) -> q(scaled)/k/v fp16 [T,128]
// block 4 waves: 64(M) x 64(N) tile, K=128 in 4 WMMA chunks
// ---------------------------------------------------------------------------
__global__ __launch_bounds__(128) void qkv_kernel(const half_t* __restrict__ xw,
                                                  const half_t* __restrict__ wt,
                                                  const float* __restrict__ bias,
                                                  half_t* __restrict__ qb,
                                                  half_t* __restrict__ kb,
                                                  half_t* __restrict__ vb) {
  int lane = threadIdx.x & 31, wave = threadIdx.x >> 5;
  int ln15 = lane & 15, lhi = lane >> 4;
  size_t row0 = (size_t)blockIdx.x * 64 + wave * 16;
  int n0 = blockIdx.y * 64;
  v16h a[4];
  #pragma unroll
  for (int kc = 0; kc < 4; ++kc)
    a[kc] = frag_load(xw + row0 * 128 + kc * 32, 128, lane);
  for (int nt = 0; nt < 4; ++nt) {
    int ng0 = n0 + nt * 16;
    v8f acc = {};
    #pragma unroll
    for (int kc = 0; kc < 4; ++kc) {
      v16h bf = frag_load(wt + (size_t)ng0 * 128 + kc * 32, 128, lane);
      acc = wmma16(a[kc], bf, acc);
    }
    int ng = ng0 + ln15;
    int which = ng >> 7, c = ng & 127;
    float bvl = bias[ng];
    half_t* dst = (which == 0) ? qb : ((which == 1) ? kb : vb);
    float scl = (which == 0) ? SCALE_Q : 1.0f;
    #pragma unroll
    for (int v = 0; v < 8; ++v) {
      size_t r = row0 + v + 8 * lhi;
      dst[r * 128 + c] = (half_t)((acc[v] + bvl) * scl);
    }
  }
}

// ---------------------------------------------------------------------------
// Kernel 3: attention per (window, head). 7 waves, one 16-row strip each.
// S = (SCALE*Q)K^T (one 16x16x32 WMMA per tile) + rel-pos bias + mask,
// shfl softmax, P fp16 in LDS, O = P x V^T (V^T staged in LDS).
// ---------------------------------------------------------------------------
__global__ __launch_bounds__(224) void attn_kernel(const half_t* __restrict__ qb,
                                                   const half_t* __restrict__ kb,
                                                   const half_t* __restrict__ vb,
                                                   const float* __restrict__ mask,
                                                   const float* __restrict__ pbt,
                                                   half_t* __restrict__ ob) {
  __shared__ alignas(16) half_t P[7][16][128];   // probs, K padded to 128
  __shared__ alignas(16) half_t Vt[32][128];     // V^T [d][j], j padded
  int wi = blockIdx.x, h = blockIdx.y;
  int tid = threadIdx.x, lane = tid & 31, wave = tid >> 5;
  int ln15 = lane & 15, lhi = lane >> 4;
  const size_t base = (size_t)wi * NTOK;

  // stage V^T
  for (int idx = tid; idx < 32 * 128; idx += 224) {
    int d = idx >> 7, j = idx & 127;
    half_t val = (half_t)0.0f;
    if (j < NTOK) val = vb[(base + j) * 128 + h * HD + d];
    Vt[d][j] = val;
  }
  __syncthreads();

  int row0 = wave * 16;
  v16h qa = frag_load_guard(qb + (base + row0) * 128 + h * HD, 128,
                            NTOK - row0, lane);
  v8f S[7];
  #pragma unroll
  for (int jt = 0; jt < 7; ++jt) {
    v16h kf = frag_load_guard(kb + (base + jt * 16) * 128 + h * HD, 128,
                              NTOK - jt * 16, lane);
    v8f z = {};
    S[jt] = wmma16(qa, kf, z);
  }

  // relative-position bias + attention mask
  const float* mrow = mask + (size_t)(wi & 255) * NTOK * NTOK;
  #pragma unroll
  for (int jt = 0; jt < 7; ++jt) {
    int j = jt * 16 + ln15;
    int jd = j / 49, jrem = j - jd * 49, jh = jrem / 7, jw = jrem - jh * 7;
    #pragma unroll
    for (int v = 0; v < 8; ++v) {
      int i = row0 + v + 8 * lhi;
      if (i < NTOK) {
        if (j < NTOK) {
          int id = i / 49, irem = i - id * 49, ih = irem / 7, iw = irem - ih * 7;
          int ridx = (id - jd + 1) * 169 + (ih - jh + 6) * 13 + (iw - jw + 6);
          S[jt][v] += pbt[ridx * HEADS + h] + mrow[i * NTOK + j];
        } else {
          S[jt][v] = -3.0e38f;   // pad columns -> exp() == 0
        }
      }
    }
  }

  // softmax per row (row lives in one 16-lane half-wave)
  #pragma unroll
  for (int v = 0; v < 8; ++v) {
    float m = -3.0e38f;
    #pragma unroll
    for (int jt = 0; jt < 7; ++jt) m = fmaxf(m, S[jt][v]);
    #pragma unroll
    for (int off = 8; off >= 1; off >>= 1) m = fmaxf(m, __shfl_xor(m, off, 32));
    float sum = 0.0f;
    #pragma unroll
    for (int jt = 0; jt < 7; ++jt) {
      float e = expf(S[jt][v] - m);
      S[jt][v] = e;
      sum += e;
    }
    #pragma unroll
    for (int off = 8; off >= 1; off >>= 1) sum += __shfl_xor(sum, off, 32);
    float inv = 1.0f / sum;
    #pragma unroll
    for (int jt = 0; jt < 7; ++jt) S[jt][v] *= inv;
  }

  // write P strip to LDS (cols 112..127 zeroed; 98..111 are already 0)
  for (int idx = lane; idx < 16 * 16; idx += 32) {
    int m = idx >> 4, c = 112 + (idx & 15);
    P[wave][m][c] = (half_t)0.0f;
  }
  #pragma unroll
  for (int jt = 0; jt < 7; ++jt)
    #pragma unroll
    for (int v = 0; v < 8; ++v)
      P[wave][v + 8 * lhi][jt * 16 + ln15] = (half_t)S[jt][v];

  // O = P x V  (K padded to 128, 4 WMMA chunks, N = 32 -> 2 tiles)
  #pragma unroll
  for (int nt = 0; nt < 2; ++nt) {
    v8f acc = {};
    #pragma unroll
    for (int kc = 0; kc < 4; ++kc) {
      v16h pa = frag_load(&P[wave][0][kc * 32], 128, lane);
      v16h vf = frag_load(&Vt[nt * 16][kc * 32], 128, lane);
      acc = wmma16(pa, vf, acc);
    }
    #pragma unroll
    for (int v = 0; v < 8; ++v) {
      int m = row0 + v + 8 * lhi;
      if (m < NTOK)
        ob[(base + m) * 128 + h * HD + nt * 16 + ln15] = (half_t)acc[v];
    }
  }
}

// ---------------------------------------------------------------------------
// Kernel 4: proj GEMM + window-reverse/roll + residual (-> d_out = x2) + LN2
// block 4 waves, each 16(M) x 128(N); LN2 row reduction via shfl
// ---------------------------------------------------------------------------
__global__ __launch_bounds__(128) void proj_kernel(const half_t* __restrict__ ao,
                                                   const half_t* __restrict__ pwt,
                                                   const float* __restrict__ pb,
                                                   const float* __restrict__ xin,
                                                   const float* __restrict__ g2,
                                                   const float* __restrict__ b2,
                                                   float* __restrict__ out,
                                                   half_t* __restrict__ xn2) {
  int lane = threadIdx.x & 31, wave = threadIdx.x >> 5;
  int ln15 = lane & 15, lhi = lane >> 4;
  size_t row0 = (size_t)blockIdx.x * 64 + wave * 16;
  v16h a[4];
  #pragma unroll
  for (int kc = 0; kc < 4; ++kc)
    a[kc] = frag_load(ao + row0 * 128 + kc * 32, 128, lane);
  v8f acc[8];
  for (int nt = 0; nt < 8; ++nt) {
    acc[nt] = (v8f){};
    #pragma unroll
    for (int kc = 0; kc < 4; ++kc) {
      v16h bf = frag_load(pwt + (size_t)(nt * 16) * 128 + kc * 32, 128, lane);
      acc[nt] = wmma16(a[kc], bf, acc[nt]);
    }
  }
  for (int v = 0; v < 8; ++v) {
    int r = (int)row0 + v + 8 * lhi;
    size_t p128 = (size_t)win_row_to_spatial(r) * 128;
    float vals[8];
    float s = 0.0f;
    #pragma unroll
    for (int nt = 0; nt < 8; ++nt) {
      int n = nt * 16 + ln15;
      float t = acc[nt][v] + pb[n] + xin[p128 + n];
      vals[nt] = t;
      s += t;
    }
    #pragma unroll
    for (int off = 8; off >= 1; off >>= 1) s += __shfl_xor(s, off, 32);
    float mu = s * (1.0f / 128.0f);
    float var = 0.0f;
    #pragma unroll
    for (int nt = 0; nt < 8; ++nt) {
      float d = vals[nt] - mu;
      var += d * d;
    }
    #pragma unroll
    for (int off = 8; off >= 1; off >>= 1) var += __shfl_xor(var, off, 32);
    float rs = rsqrtf(var * (1.0f / 128.0f) + 1e-5f);
    #pragma unroll
    for (int nt = 0; nt < 8; ++nt) {
      int n = nt * 16 + ln15;
      out[p128 + n] = vals[nt];
      xn2[p128 + n] = (half_t)((vals[nt] - mu) * rs * g2[n] + b2[n]);
    }
  }
}

// ---------------------------------------------------------------------------
// Kernel 5: fused MLP. FC1+bias+GELU(exact erf) -> 64x512 fp16 strip in LDS,
// then FC2+bias+residual into d_out. Hidden never touches HBM.
// ---------------------------------------------------------------------------
__global__ __launch_bounds__(128) void mlp_kernel(const half_t* __restrict__ xn2,
                                                  const half_t* __restrict__ w1t,
                                                  const float* __restrict__ b1,
                                                  const half_t* __restrict__ w2t,
                                                  const float* __restrict__ b2v,
                                                  float* __restrict__ out) {
  __shared__ alignas(16) half_t Hs[64][512];
  int lane = threadIdx.x & 31, wave = threadIdx.x >> 5;
  int ln15 = lane & 15, lhi = lane >> 4;
  size_t row0 = (size_t)blockIdx.x * 64 + wave * 16;
  v16h a[4];
  #pragma unroll
  for (int kc = 0; kc < 4; ++kc)
    a[kc] = frag_load(xn2 + row0 * 128 + kc * 32, 128, lane);
  for (int nt = 0; nt < 32; ++nt) {
    v8f acc = {};
    #pragma unroll
    for (int kc = 0; kc < 4; ++kc) {
      v16h bf = frag_load(w1t + (size_t)(nt * 16) * 128 + kc * 32, 128, lane);
      acc = wmma16(a[kc], bf, acc);
    }
    int n = nt * 16 + ln15;
    float bb = b1[n];
    #pragma unroll
    for (int v = 0; v < 8; ++v) {
      float t = acc[v] + bb;
      t = 0.5f * t * (1.0f + erff(t * 0.7071067811865476f));
      Hs[wave * 16 + v + 8 * lhi][n] = (half_t)t;
    }
  }
  __syncthreads();
  for (int nt = 0; nt < 8; ++nt) {
    v8f acc = {};
    for (int kc = 0; kc < 16; ++kc) {
      v16h pa = frag_load(&Hs[wave * 16][kc * 32], 512, lane);
      v16h bf = frag_load(w2t + (size_t)(nt * 16) * 512 + kc * 32, 512, lane);
      acc = wmma16(pa, bf, acc);
    }
    int n = nt * 16 + ln15;
    float bb = b2v[n];
    #pragma unroll
    for (int v = 0; v < 8; ++v) {
      size_t r = row0 + v + 8 * lhi;
      out[r * 128 + n] += acc[v] + bb;
    }
  }
}

// ---------------------------------------------------------------------------
extern "C" void kernel_launch(void* const* d_in, const int* in_sizes, int n_in,
                              void* d_out, int out_size, void* d_ws, size_t ws_size,
                              hipStream_t stream) {
  const float* x     = (const float*)d_in[0];
  const float* mask  = (const float*)d_in[1];
  const float* n1g   = (const float*)d_in[2];
  const float* n1b   = (const float*)d_in[3];
  const float* qkvw  = (const float*)d_in[4];
  const float* qkvb  = (const float*)d_in[5];
  const float* pbt   = (const float*)d_in[6];
  const float* projw = (const float*)d_in[7];
  const float* projb = (const float*)d_in[8];
  const float* n2g   = (const float*)d_in[9];
  const float* n2b   = (const float*)d_in[10];
  const float* w1    = (const float*)d_in[11];
  const float* b1    = (const float*)d_in[12];
  const float* w2    = (const float*)d_in[13];
  const float* b2    = (const float*)d_in[14];
  float* out = (float*)d_out;

  char* ws = (char*)d_ws;
  const size_t SZ = (size_t)T_TOKENS * 128 * sizeof(half_t);   // 51,380,224 B
  half_t* xw     = (half_t*)(ws);            // later reused as attention output
  half_t* qb     = (half_t*)(ws + SZ);       // later reused as xn2
  half_t* kb     = (half_t*)(ws + 2 * SZ);
  half_t* vb     = (half_t*)(ws + 3 * SZ);
  half_t* qkvwt  = (half_t*)(ws + 4 * SZ);
  half_t* projwt = (half_t*)(ws + 4 * SZ + 98304);
  half_t* w1t    = (half_t*)(ws + 4 * SZ + 98304 + 32768);
  half_t* w2t    = (half_t*)(ws + 4 * SZ + 98304 + 32768 + 131072);
  half_t* ao  = xw;   // attention output aliases xw (xw dead after QKV GEMM)
  half_t* xn2 = qb;   // LN2 output aliases q (q dead after attention)

  // 0) weights -> fp16, transposed [N][K]
  wprep_kernel<<<(128 * 384 + 255) / 256, 256, 0, stream>>>(qkvw, qkvwt, 128, 384);
  wprep_kernel<<<(128 * 128 + 255) / 256, 256, 0, stream>>>(projw, projwt, 128, 128);
  wprep_kernel<<<(128 * 512 + 255) / 256, 256, 0, stream>>>(w1, w1t, 128, 512);
  wprep_kernel<<<(512 * 128 + 255) / 256, 256, 0, stream>>>(w2, w2t, 512, 128);
  // 1) LN1 + shift + window partition
  ln1_kernel<<<T_TOKENS / 8, 256, 0, stream>>>(x, n1g, n1b, xw);
  // 2) QKV
  qkv_kernel<<<dim3(T_TOKENS / 64, 6), 128, 0, stream>>>(xw, qkvwt, qkvb, qb, kb, vb);
  // 3) windowed attention
  attn_kernel<<<dim3(NWIN, HEADS), 224, 0, stream>>>(qb, kb, vb, mask, pbt, ao);
  // 4) proj + reverse/roll + residual -> d_out(x2), LN2 -> xn2
  proj_kernel<<<T_TOKENS / 64, 128, 0, stream>>>(ao, projwt, projb, x, n2g, n2b, out, xn2);
  // 5) fused MLP + residual into d_out
  mlp_kernel<<<T_TOKENS / 64, 128, 0, stream>>>(xn2, w1t, b1, w2t, b2, out);
}
